// JetClassifierGNN_47218870452627
// MI455X (gfx1250) — compile-verified, matrix operand order checked by reading, not run on hardware
//
#include <hip/hip_runtime.h>

typedef __attribute__((ext_vector_type(16))) _Float16 v16h;
typedef __attribute__((ext_vector_type(8)))  _Float16 v8h;
typedef __attribute__((ext_vector_type(4)))  _Float16 v4h;
typedef __attribute__((ext_vector_type(4)))  float    v4f;
typedef __attribute__((ext_vector_type(8)))  float    v8f;

#define IN_DIM 16
#define HID 128
#define OUT_DIM 2

// ---------------- zero workspace ----------------
__global__ void k_zero(float* __restrict__ p, long long n) {
    long long i = (long long)blockIdx.x * blockDim.x + threadIdx.x;
    long long stride = (long long)gridDim.x * blockDim.x;
    for (; i < n; i += stride) p[i] = 0.0f;
}

// ---- pre-pack weights into f16 WMMA B-fragment order (once per launch) ----
// pB1: [colTile(8)][lane(32)][elem(16)]            (layer-1, K=32)
// pB2: [ks(8)][colTile(8)][lane(32)][elem(16)]     (layer-2, K=256)
__global__ void k_pack_w(const float* __restrict__ W1_rel,
                         const float* __restrict__ W1_root,
                         const float* __restrict__ W2_rel,
                         const float* __restrict__ W2_root,
                         _Float16* __restrict__ pB1,
                         _Float16* __restrict__ pB2) {
    int t = blockIdx.x * blockDim.x + threadIdx.x;
    if (t < 8 * 32 * 16) {                 // layer-1: 4096 elements
        int colTile = t >> 9;
        int lane    = (t >> 4) & 31;
        int elem    = t & 15;
        int v = elem >> 1, j = elem & 1;
        int hk   = (lane < 16) ? 0 : 8;
        int base = (v < 4) ? (2 * v) : (2 * (v - 4) + 16);
        int K    = base + hk + j;          // 0..31
        int col  = colTile * 16 + (lane & 15);
        float w  = (K < IN_DIM) ? W1_rel[K * HID + col]
                                : W1_root[(K - IN_DIM) * HID + col];
        pB1[t] = (_Float16)w;
    }
    if (t < 8 * 8 * 32 * 16) {             // layer-2: 32768 elements
        int ks      = t >> 12;
        int rem     = t & 4095;
        int colTile = rem >> 9;
        int lane    = (rem >> 4) & 31;
        int elem    = rem & 15;
        int v = elem >> 1, j = elem & 1;
        int hk   = (lane < 16) ? 0 : 8;
        int base = (v < 4) ? (2 * v) : (2 * (v - 4) + 16);
        int Kg   = ks * 32 + base + hk + j;   // 0..255
        int col  = colTile * 16 + (lane & 15);
        float w  = (Kg < HID) ? W2_rel[Kg * HID + col]
                              : W2_root[(Kg - HID) * HID + col];
        pB2[t] = (_Float16)w;
    }
}

// ---------------- layer-1 edge aggregation: agg1[dst] += x[src] ----------------
__global__ void k_agg1(const float* __restrict__ x,
                       const long long* __restrict__ src,
                       const long long* __restrict__ dst,
                       float* __restrict__ agg1, int E) {
    int t = blockIdx.x * blockDim.x + threadIdx.x;
    int e = t >> 4;
    int d = t & 15;
    if (e >= E) return;
    int s  = (int)src[e];
    int dd = (int)dst[e];
    atomicAdd(&agg1[dd * IN_DIM + d], x[s * IN_DIM + d]);
}

// ---------------- layer 1: h1 = relu([agg1|x] @ [W1_rel;W1_root] + b1), K=32 ----
__global__ void k_layer1(const float* __restrict__ x,
                         const float* __restrict__ agg1,
                         const _Float16* __restrict__ pB1,
                         const float* __restrict__ b1,
                         _Float16* __restrict__ h1, int N) {
    int lane    = threadIdx.x & 31;
    int wave    = threadIdx.x >> 5;      // colTile 0..7
    int rowTile = blockIdx.x;
    int col     = wave * 16 + (lane & 15);
    int hk      = (lane < 16) ? 0 : 8;
    int node    = rowTile * 16 + (lane & 15);
    bool full   = (rowTile * 16 + 16 <= N);  // uniform per block
    int nodeC   = (node < N) ? node : (N - 1);

    // A fragment: elems 0..7 = agg1[node][hk..hk+7], 8..15 = x[node][hk..hk+7]
    v4f p0 = *(const v4f*)(agg1 + (size_t)nodeC * IN_DIM + hk);
    v4f p1 = *(const v4f*)(agg1 + (size_t)nodeC * IN_DIM + hk + 4);
    v4f p2 = *(const v4f*)(x    + (size_t)nodeC * IN_DIM + hk);
    v4f p3 = *(const v4f*)(x    + (size_t)nodeC * IN_DIM + hk + 4);
    v16h a;
#pragma unroll
    for (int i = 0; i < 4; ++i) {
        a[i]      = (_Float16)p0[i];
        a[4 + i]  = (_Float16)p1[i];
        a[8 + i]  = (_Float16)p2[i];
        a[12 + i] = (_Float16)p3[i];
    }
    // B fragment: one coalesced 32B load per lane
    v16h b = *(const v16h*)(pB1 + (((size_t)wave * 32 + lane) << 4));

    v8f c = {};
    c = __builtin_amdgcn_wmma_f32_16x16x32_f16(false, a, false, b,
                                               (short)0, c, false, false);

    float bias = b1[col];
    if (full) {
#pragma unroll
        for (int r = 0; r < 8; ++r) {
            int nodeR = rowTile * 16 + r + ((lane < 16) ? 0 : 8);
            h1[(size_t)nodeR * HID + col] = (_Float16)fmaxf(c[r] + bias, 0.0f);
        }
    } else {
#pragma unroll
        for (int r = 0; r < 8; ++r) {
            int nodeR = rowTile * 16 + r + ((lane < 16) ? 0 : 8);
            if (nodeR < N)
                h1[(size_t)nodeR * HID + col] = (_Float16)fmaxf(c[r] + bias, 0.0f);
        }
    }
}

// ---------------- layer-2 edge aggregation: agg2[dst] += h1[src] (f16 gather) ---
__global__ void k_agg2(const _Float16* __restrict__ h1,
                       const long long* __restrict__ src,
                       const long long* __restrict__ dst,
                       float* __restrict__ agg2, int E) {
    int e    = blockIdx.x * 8 + (threadIdx.x >> 5);   // warp per edge
    int lane = threadIdx.x & 31;                      // 4 feats per lane
    if (e >= E) return;
    int s  = (int)src[e];
    int dd = (int)dst[e];
    v4h hv = *(const v4h*)(h1 + (size_t)s * HID + lane * 4);
    float* ag = agg2 + (size_t)dd * HID + lane * 4;
    atomicAdd(ag + 0, (float)hv[0]);
    atomicAdd(ag + 1, (float)hv[1]);
    atomicAdd(ag + 2, (float)hv[2]);
    atomicAdd(ag + 3, (float)hv[3]);
}

// ------- layer 2 + pool: h2 = relu([agg2|h1] @ [W2_rel;W2_root] + b2), K=256 ----
__global__ void k_layer2_pool(const _Float16* __restrict__ h1,
                              const float* __restrict__ agg2,
                              const _Float16* __restrict__ pB2,
                              const float* __restrict__ b2,
                              const long long* __restrict__ batch,
                              float* __restrict__ sums, int N) {
    int lane    = threadIdx.x & 31;
    int wave    = threadIdx.x >> 5;      // colTile 0..7
    int rowTile = blockIdx.x;
    int col     = wave * 16 + (lane & 15);
    int hk      = (lane < 16) ? 0 : 8;
    int node    = rowTile * 16 + (lane & 15);
    bool full   = (rowTile * 16 + 16 <= N);
    int nodeC   = (node < N) ? node : (N - 1);

    v8f c = {};
#pragma unroll
    for (int ks = 0; ks < 8; ++ks) {     // 8 x K=32 chained WMMAs -> K=256
        v16h a;
        if (ks < 4) {
            // A from agg2 (f32): two contiguous 8-float chunks per lane
            const float* arow = agg2 + (size_t)nodeC * HID + ks * 32 + hk;
            v4f q0 = *(const v4f*)(arow);
            v4f q1 = *(const v4f*)(arow + 4);
            v4f q2 = *(const v4f*)(arow + 16);
            v4f q3 = *(const v4f*)(arow + 20);
#pragma unroll
            for (int i = 0; i < 4; ++i) {
                a[i]      = (_Float16)q0[i];
                a[4 + i]  = (_Float16)q1[i];
                a[8 + i]  = (_Float16)q2[i];
                a[12 + i] = (_Float16)q3[i];
            }
        } else {
            // A from h1 (f16): two contiguous 16B chunks, no conversion
            const _Float16* hrow = h1 + (size_t)nodeC * HID + (ks - 4) * 32 + hk;
            v8h q0 = *(const v8h*)(hrow);
            v8h q1 = *(const v8h*)(hrow + 16);
            a = __builtin_shufflevector(q0, q1, 0, 1, 2, 3, 4, 5, 6, 7,
                                        8, 9, 10, 11, 12, 13, 14, 15);
        }
        v16h b = *(const v16h*)(pB2 + ((((size_t)ks * 8 + wave) * 32 + lane) << 4));
        c = __builtin_amdgcn_wmma_f32_16x16x32_f16(false, a, false, b,
                                                   (short)0, c, false, false);
    }

    float bias = b2[col];
    if (full) {
#pragma unroll
        for (int r = 0; r < 8; ++r) {
            int nodeR = rowTile * 16 + r + ((lane < 16) ? 0 : 8);
            float val = fmaxf(c[r] + bias, 0.0f);
            int g = (int)batch[nodeR];
            atomicAdd(&sums[(size_t)g * HID + col], val);
        }
    } else {
#pragma unroll
        for (int r = 0; r < 8; ++r) {
            int nodeR = rowTile * 16 + r + ((lane < 16) ? 0 : 8);
            if (nodeR < N) {
                float val = fmaxf(c[r] + bias, 0.0f);
                int g = (int)batch[nodeR];
                atomicAdd(&sums[(size_t)g * HID + col], val);
            }
        }
    }
}

// ---------------- per-graph node counts ----------------
__global__ void k_count(const long long* __restrict__ batch,
                        float* __restrict__ cnt, int N) {
    int i = blockIdx.x * blockDim.x + threadIdx.x;
    if (i < N) atomicAdd(&cnt[(int)batch[i]], 1.0f);
}

// ---------------- head: out = (sums / max(cnt,1)) @ Wfc + bfc ----------------
__global__ void k_head(const float* __restrict__ sums,
                       const float* __restrict__ cnt,
                       const float* __restrict__ Wfc,
                       const float* __restrict__ bfc,
                       float* __restrict__ out, int G) {
    int t = blockIdx.x * blockDim.x + threadIdx.x;
    int g = t >> 1;
    int o = t & 1;
    if (g >= G) return;
    float inv = 1.0f / fmaxf(cnt[g], 1.0f);
    float acc = 0.0f;
#pragma unroll 8
    for (int k = 0; k < HID; ++k)
        acc += sums[(size_t)g * HID + k] * Wfc[k * OUT_DIM + o];
    out[g * OUT_DIM + o] = acc * inv + bfc[o];
}

extern "C" void kernel_launch(void* const* d_in, const int* in_sizes, int n_in,
                              void* d_out, int out_size, void* d_ws, size_t ws_size,
                              hipStream_t stream) {
    const float*     x       = (const float*)d_in[0];
    const long long* eidx    = (const long long*)d_in[1];  // int64 per reference
    const long long* batch   = (const long long*)d_in[2];
    const float*     W1_rel  = (const float*)d_in[3];
    const float*     b1      = (const float*)d_in[4];
    const float*     W1_root = (const float*)d_in[5];
    const float*     W2_rel  = (const float*)d_in[6];
    const float*     b2      = (const float*)d_in[7];
    const float*     W2_root = (const float*)d_in[8];
    const float*     Wfc     = (const float*)d_in[9];
    const float*     bfc     = (const float*)d_in[10];
    float*           out     = (float*)d_out;

    const int N = in_sizes[0] / IN_DIM;
    const int E = in_sizes[1] / 2;
    const int G = out_size / OUT_DIM;
    const long long* src = eidx;
    const long long* dst = eidx + E;

    // ---- workspace layout (256B-aligned slabs) ----
    char* base = (char*)d_ws;
    size_t off = 0;
    auto alignUp = [](size_t v) { return (v + 255) & ~(size_t)255; };
    float* agg1 = (float*)(base + off); off += alignUp((size_t)N * IN_DIM * 4);
    float* agg2 = (float*)(base + off); off += alignUp((size_t)N * HID * 4);
    float* sums = (float*)(base + off); off += alignUp((size_t)G * HID * 4);
    float* cnt  = (float*)(base + off); off += alignUp((size_t)G * 4);
    _Float16* h1  = (_Float16*)(base + off); off += alignUp((size_t)N * HID * 2);
    _Float16* pB1 = (_Float16*)(base + off); off += alignUp((size_t)8 * 32 * 16 * 2);
    _Float16* pB2 = (_Float16*)(base + off); off += alignUp((size_t)8 * 8 * 32 * 16 * 2);
    (void)ws_size; (void)n_in;

    // zero the accumulator region [agg1 .. cnt] (contiguous prefix of ws)
    long long zn = (long long)(((char*)h1 - base) / 4);
    k_zero<<<2048, 256, 0, stream>>>((float*)base, zn);

    // pre-pack weights into WMMA fragment layout (f16)
    k_pack_w<<<(8 * 8 * 32 * 16 + 255) / 256, 256, 0, stream>>>(
        W1_rel, W1_root, W2_rel, W2_root, pB1, pB2);

    // layer-1 aggregation: thread per (edge, feature)
    long long t1 = (long long)E * IN_DIM;
    k_agg1<<<(int)((t1 + 255) / 256), 256, 0, stream>>>(x, src, dst, agg1, E);

    // layer-1 WMMA GEMM (row tiles x 8 col-tile waves)
    int rowTiles = (N + 15) / 16;
    k_layer1<<<rowTiles, 256, 0, stream>>>(x, agg1, pB1, b1, h1, N);

    // layer-2 aggregation: warp per edge
    k_agg2<<<(E + 7) / 8, 256, 0, stream>>>(h1, src, dst, agg2, E);

    // layer-2 WMMA GEMM fused with mean-pool scatter
    k_layer2_pool<<<rowTiles, 256, 0, stream>>>(h1, agg2, pB2, b2, batch, sums, N);

    // per-graph counts
    k_count<<<(N + 255) / 256, 256, 0, stream>>>(batch, cnt, N);

    // head
    k_head<<<(G * OUT_DIM + 255) / 256, 256, 0, stream>>>(sums, cnt, Wfc, bfc, out, G);
}
